// DirectSelfAttention_36309653520666
// MI455X (gfx1250) — matrix operand order repaired and checked by reading, compiled, not run on hardware
//
#include <hip/hip_runtime.h>
#include <math.h>

// ---------------------------------------------------------------------------
// MI455X (gfx1250) transformer forward.
// All matmuls -> v_wmma_f32_16x16x32_bf16 (wave32). Workload is HBM-bound
// (~620 GFLOP vs ~0.7 GB traffic @ 23.3 TB/s), so bf16 operands + fused
// epilogues + flash-attention streaming, and CDNA5 async global->LDS copies
// (ASYNCcnt) to overlap staging with WMMA compute (double-buffered GEMM).
// ---------------------------------------------------------------------------

typedef __attribute__((ext_vector_type(16))) __bf16 bf16x16;
typedef __attribute__((ext_vector_type(8)))  float  floatx8;
typedef __attribute__((ext_vector_type(4)))  float  floatx4;
typedef __attribute__((ext_vector_type(4)))  unsigned short ushortx4;
typedef unsigned short ushort_t;

#define DIMD   1024
#define INNERD 512
#define FFD    4096
#define NTOK   4096   /* 2 * 2048 tokens */
#define SEQ    2048
#define HEADSD 8
#define DHEAD  64

__device__ __forceinline__ ushort_t f2bf(float f) {
  unsigned int u = __float_as_uint(f);
  u += 0x7FFFu + ((u >> 16) & 1u);          // round-to-nearest-even
  return (ushort_t)(u >> 16);
}

__device__ __forceinline__ bf16x16 frag_from(const ushort_t* p0, const ushort_t* p1) {
  bf16x16 r;
  ((uint4*)&r)[0] = *(const uint4*)p0;
  ((uint4*)&r)[1] = *(const uint4*)p1;
  return r;
}

// CDNA5 async global->LDS copy (16B per lane), tracked by ASYNCcnt.
// lds_off = low 32 bits of the flat shared address (= LDS byte address).
__device__ __forceinline__ void async_copy_b128(unsigned lds_off, const ushort_t* g) {
  asm volatile("global_load_async_to_lds_b128 %0, %1, off"
               :: "v"(lds_off), "v"(g) : "memory");
}
#define S_WAIT_ASYNCCNT(N) asm volatile("s_wait_asynccnt %0" :: "i"(N) : "memory")

__device__ __forceinline__ unsigned lds_addr(const void* p) {
  return (unsigned)(size_t)p;   // flat shared addr low 32 bits = LDS offset
}

// ---------------------------------------------------------------------------
// GEMM: C[M,N] = act(A_bf16[M,K] @ B_bf16[K,N] + bias) (+ residual)
// flags: 1=bias, 2=gelu(erf), 4=residual(f32), 8=bf16 output (else f32)
// Block 256 thr = 8 waves; block tile 128x128, BK=32; wave tile 32x64.
// Double-buffered LDS; A tile via async-to-LDS, B tile prefetched in VGPRs.
// ---------------------------------------------------------------------------
#define BM 128
#define BN 128
#define BK 32
#define LDS_LD 40   /* padded row stride (bf16 elems): banks stay conflict-free */

__global__ __launch_bounds__(256) void gemm_bf16_kernel(
    const ushort_t* __restrict__ A, const ushort_t* __restrict__ B,
    const float* __restrict__ bias, const float* __restrict__ resid,
    float* __restrict__ outF, ushort_t* __restrict__ outB,
    int M, int N, int K, int flags)
{
  __shared__ __align__(16) ushort_t As[2][BM * LDS_LD];
  __shared__ __align__(16) ushort_t Bs[2][BN * LDS_LD];

  const int tid  = threadIdx.x;
  const int lane = tid & 31;
  const int wave = tid >> 5;
  const int half = lane >> 4;
  const int l15  = lane & 15;
  const int wm   = wave >> 1;           // 0..3
  const int wn   = wave & 1;            // 0..1
  const int m0   = blockIdx.y * BM;
  const int n0   = blockIdx.x * BN;

  const floatx8 vzero = {};
  floatx8 acc[2][4];
#pragma unroll
  for (int i = 0; i < 2; ++i)
#pragma unroll
    for (int j = 0; j < 4; ++j) acc[i][j] = vzero;

  const int ar = tid >> 1;              // A tile: 128 rows, 2 thr/row
  const int ac = (tid & 1) * 16;        // 16 bf16 each (2 x b128)
  const int bk = tid & 31;              // B tile: one k, 16 n's per thread
  const int bn = (tid >> 5) * 16;

  const ushort_t* Arow = A + (size_t)(m0 + ar) * K + ac;
  const ushort_t* Brow = B + (size_t)bk * N + n0 + bn;
  const unsigned aoff[2] = { lds_addr(&As[0][ar * LDS_LD + ac]),
                             lds_addr(&As[1][ar * LDS_LD + ac]) };

  union BReg { uint4 v[2]; ushort_t s[16]; };
  BReg bu, bu2;

  // prologue: stage tile 0
  async_copy_b128(aoff[0],      Arow);
  async_copy_b128(aoff[0] + 16, Arow + 8);
  bu.v[0] = *(const uint4*)Brow;
  bu.v[1] = *(const uint4*)(Brow + 8);

  const int T = K / BK;
  for (int kt = 0; kt < T; ++kt) {
    const int cur = kt & 1, nxt = cur ^ 1;

    // scatter prefetched B tile -> Bs[cur][n][k] (transposed)
#pragma unroll
    for (int i = 0; i < 16; ++i) Bs[cur][(bn + i) * LDS_LD + bk] = bu.s[i];

    const bool hasNext = (kt + 1) < T;
    if (hasNext) {
      const ushort_t* an = Arow + (size_t)(kt + 1) * BK;
      async_copy_b128(aoff[nxt],      an);
      async_copy_b128(aoff[nxt] + 16, an + 8);
      const ushort_t* bnp = Brow + (size_t)(kt + 1) * BK * N;
      bu2.v[0] = *(const uint4*)bnp;
      bu2.v[1] = *(const uint4*)(bnp + 8);
    }
    // in-order ASYNC completion: <=2 outstanding -> this tile's chunks landed
    if (hasNext) { S_WAIT_ASYNCCNT(2); } else { S_WAIT_ASYNCCNT(0); }
    __syncthreads();

    bf16x16 af[2], bfr[4];
#pragma unroll
    for (int tm = 0; tm < 2; ++tm) {
      const ushort_t* p = &As[cur][(wm * 32 + tm * 16 + l15) * LDS_LD];
      af[tm] = frag_from(p + half * 8, p + 16 + half * 8);
    }
#pragma unroll
    for (int tn = 0; tn < 4; ++tn) {
      const ushort_t* p = &Bs[cur][(wn * 64 + tn * 16 + l15) * LDS_LD + half * 8];
      bfr[tn] = frag_from(p, p + 16);
    }
#pragma unroll
    for (int tm = 0; tm < 2; ++tm)
#pragma unroll
      for (int tn = 0; tn < 4; ++tn)
        acc[tm][tn] = __builtin_amdgcn_wmma_f32_16x16x32_bf16(
            false, af[tm], false, bfr[tn], (short)0, acc[tm][tn], false, false);
    __syncthreads();   // protect As[cur]/Bs[cur] before next overwrite cycle

    bu = bu2;
  }

  // epilogue: D layout row = v + 8*half, col = lane&15
#pragma unroll
  for (int tm = 0; tm < 2; ++tm) {
    const int rbase = m0 + wm * 32 + tm * 16 + half * 8;
#pragma unroll
    for (int tn = 0; tn < 4; ++tn) {
      const int col = n0 + wn * 64 + tn * 16 + l15;
      const float bval = (flags & 1) ? bias[col] : 0.f;
#pragma unroll
      for (int v = 0; v < 8; ++v) {
        const int row = rbase + v;
        float c = acc[tm][tn][v] + bval;
        if (flags & 2) c = 0.5f * c * (1.f + erff(c * 0.70710678118654752f));
        if (flags & 4) c += resid[(size_t)row * N + col];
        if (flags & 8) outB[(size_t)row * N + col] = f2bf(c);
        else           outF[(size_t)row * N + col] = c;
      }
    }
  }
}

// ---------------------------------------------------------------------------
// Flash attention: block = 8 waves x 16 query rows (128 rows), one (b,h).
// j streamed in tiles of 32; online softmax; K tile via async-to-LDS.
// ---------------------------------------------------------------------------
#define KS_LD 72
#define VT_LD 40
#define PS_LD 40

__global__ __launch_bounds__(256) void attn_kernel(
    const ushort_t* __restrict__ Qb, const ushort_t* __restrict__ Kb,
    const ushort_t* __restrict__ Vb, ushort_t* __restrict__ outB)
{
  __shared__ __align__(16) ushort_t Ks[32 * KS_LD];      // K tile [j][d]
  __shared__ __align__(16) ushort_t Vt[64 * VT_LD];      // V tile transposed [d][j]
  __shared__ __align__(16) ushort_t Ps[8][16 * PS_LD];   // per-wave P staging

  const int tid  = threadIdx.x;
  const int lane = tid & 31;
  const int wave = tid >> 5;
  const int half = lane >> 4;
  const int l15  = lane & 15;

  const int bh = blockIdx.y;
  const int b  = bh >> 3;
  const int h  = bh & 7;
  const ushort_t* Qh = Qb + (size_t)bh * SEQ * DHEAD;
  const ushort_t* Kh = Kb + (size_t)bh * SEQ * DHEAD;
  const ushort_t* Vh = Vb + (size_t)bh * SEQ * DHEAD;

  const int q0 = blockIdx.x * 128 + wave * 16;

  bf16x16 qf[2];                       // Q 16x64 -> two k-steps of 32
#pragma unroll
  for (int s = 0; s < 2; ++s) {
    const ushort_t* p = Qh + (size_t)(q0 + l15) * DHEAD + s * 32 + half * 8;
    qf[s] = frag_from(p, p + 16);
  }

  const floatx8 vzero = {};
  floatx8 accO[4];
#pragma unroll
  for (int i = 0; i < 4; ++i) accO[i] = vzero;
  float mrow[8], lrow[8];
#pragma unroll
  for (int v = 0; v < 8; ++v) { mrow[v] = -3.0e38f; lrow[v] = 0.f; }

  const int kr = tid >> 3, kc = (tid & 7) * 8;
  const int vj = tid & 31, vd = (tid >> 5) * 8;
  const unsigned ksoff = lds_addr(&Ks[kr * KS_LD + kc]);

  for (int j0 = 0; j0 < SEQ; j0 += 32) {
    // K tile: async global->LDS (in flight while V is staged below)
    async_copy_b128(ksoff, Kh + (size_t)(j0 + kr) * DHEAD + kc);
    {
      union { uint4 v; ushort_t s[8]; } u;
      u.v = *(const uint4*)(Vh + (size_t)(j0 + vj) * DHEAD + vd);
#pragma unroll
      for (int i = 0; i < 8; ++i) Vt[(vd + i) * VT_LD + vj] = u.s[i];
    }
    S_WAIT_ASYNCCNT(0);
    __syncthreads();

    floatx8 sAcc[2];
#pragma unroll
    for (int jn = 0; jn < 2; ++jn) {
      sAcc[jn] = vzero;
#pragma unroll
      for (int s = 0; s < 2; ++s) {
        const ushort_t* p = &Ks[(jn * 16 + l15) * KS_LD + s * 32 + half * 8];
        bf16x16 kfrag = frag_from(p, p + 16);
        sAcc[jn] = __builtin_amdgcn_wmma_f32_16x16x32_bf16(
            false, qf[s], false, kfrag, (short)0, sAcc[jn], false, false);
      }
    }

    float p0[8], p1[8], scl[8];
#pragma unroll
    for (int v = 0; v < 8; ++v) {
      float tm = fmaxf(sAcc[0][v], sAcc[1][v]);
      tm = fmaxf(tm, __shfl_xor(tm, 1, 32));
      tm = fmaxf(tm, __shfl_xor(tm, 2, 32));
      tm = fmaxf(tm, __shfl_xor(tm, 4, 32));
      tm = fmaxf(tm, __shfl_xor(tm, 8, 32));
      const float mn = fmaxf(mrow[v], tm);
      scl[v] = __expf(mrow[v] - mn);
      p0[v]  = __expf(sAcc[0][v] - mn);
      p1[v]  = __expf(sAcc[1][v] - mn);
      float rs = p0[v] + p1[v];
      rs += __shfl_xor(rs, 1, 32);
      rs += __shfl_xor(rs, 2, 32);
      rs += __shfl_xor(rs, 4, 32);
      rs += __shfl_xor(rs, 8, 32);
      lrow[v] = lrow[v] * scl[v] + rs;
      mrow[v] = mn;
    }
#pragma unroll
    for (int nd = 0; nd < 4; ++nd)
#pragma unroll
      for (int v = 0; v < 8; ++v) accO[nd][v] *= scl[v];

    // stage P (D-layout) -> LDS -> reload as A-fragment (wave-local)
#pragma unroll
    for (int v = 0; v < 8; ++v) {
      const int pr = (v + 8 * half) * PS_LD;
      Ps[wave][pr + l15]      = f2bf(p0[v]);
      Ps[wave][pr + 16 + l15] = f2bf(p1[v]);
    }
    bf16x16 pf;
    {
      const ushort_t* p = &Ps[wave][l15 * PS_LD];
      pf = frag_from(p + half * 8, p + 16 + half * 8);
    }
#pragma unroll
    for (int nd = 0; nd < 4; ++nd) {
      const ushort_t* p = &Vt[(nd * 16 + l15) * VT_LD + half * 8];
      bf16x16 vfrag = frag_from(p, p + 16);
      accO[nd] = __builtin_amdgcn_wmma_f32_16x16x32_bf16(
          false, pf, false, vfrag, (short)0, accO[nd], false, false);
    }
    __syncthreads();
  }

#pragma unroll
  for (int v = 0; v < 8; ++v) {
    const float inv = 1.f / lrow[v];
    const size_t tok = (size_t)b * SEQ + q0 + v + 8 * half;
#pragma unroll
    for (int nd = 0; nd < 4; ++nd) {
      const int col = h * DHEAD + nd * 16 + l15;
      outB[tok * INNERD + col] = f2bf(accO[nd][v] * inv);
    }
  }
}

// ---------------------------------------------------------------------------
// LayerNorm: one block per 1024-wide row. bf16out=1 -> outB, else outF.
// ---------------------------------------------------------------------------
__global__ __launch_bounds__(256) void ln_kernel(
    const float* __restrict__ x, const float* __restrict__ g,
    const float* __restrict__ bta, ushort_t* __restrict__ outB,
    float* __restrict__ outF, int bf16out)
{
  __shared__ float s1[256], s2[256];
  const int row = blockIdx.x;
  const int tid = threadIdx.x;
  const float4 v = *(const float4*)(x + (size_t)row * DIMD + tid * 4);
  s1[tid] = v.x + v.y + v.z + v.w;
  s2[tid] = v.x*v.x + v.y*v.y + v.z*v.z + v.w*v.w;
  __syncthreads();
  for (int off = 128; off > 0; off >>= 1) {
    if (tid < off) { s1[tid] += s1[tid+off]; s2[tid] += s2[tid+off]; }
    __syncthreads();
  }
  const float mean = s1[0] * (1.f / DIMD);
  const float var  = s2[0] * (1.f / DIMD) - mean * mean;
  const float rs   = rsqrtf(var + 1e-5f);
  const float xv[4] = {v.x, v.y, v.z, v.w};
#pragma unroll
  for (int i = 0; i < 4; ++i) {
    const int c = tid * 4 + i;
    const float y = (xv[i] - mean) * rs * g[c] + bta[c];
    if (bf16out) outB[(size_t)row * DIMD + c] = f2bf(y);
    else         outF[(size_t)row * DIMD + c] = y;
  }
}

// f32 -> bf16, 4 elems/thread; weights are streamed once -> non-temporal
// (keep the 192MB L2 for reused activations). Uses clang ext-vector types,
// which __builtin_nontemporal_* requires.
__global__ __launch_bounds__(256) void cvt_kernel(
    const float* __restrict__ in, ushort_t* __restrict__ out, int n4)
{
  const int i = blockIdx.x * 256 + threadIdx.x;
  if (i < n4) {
    const floatx4 v = __builtin_nontemporal_load(((const floatx4*)in) + i);
    ushortx4 o;
    o.x = f2bf(v.x); o.y = f2bf(v.y); o.z = f2bf(v.z); o.w = f2bf(v.w);
    __builtin_nontemporal_store(o, ((ushortx4*)out) + i);
  }
}

// qkv f32 [tok][1536] -> Qb/Kb/Vb bf16 [b][h][n][d]; Q pre-scaled by d^-1/2
__global__ __launch_bounds__(256) void repack_qkv_kernel(
    const float* __restrict__ qkv, ushort_t* __restrict__ Qb,
    ushort_t* __restrict__ Kb, ushort_t* __restrict__ Vb)
{
  const int idx = blockIdx.x * 256 + threadIdx.x;   // < NTOK*1536
  const int t = idx / 1536;
  const int c = idx - t * 1536;
  const int b = t >> 11;
  const int n = t & 2047;
  const int which = c >> 9;
  const int r = c & 511;
  const int h = r >> 6;
  const int d = r & 63;
  const float v = qkv[(size_t)t * 1536 + c];
  const size_t dst = ((size_t)(b * HEADSD + h) * SEQ + n) * DHEAD + d;
  if (which == 0)      Qb[dst] = f2bf(v * 0.125f);
  else if (which == 1) Kb[dst] = f2bf(v);
  else                 Vb[dst] = f2bf(v);
}

// ---------------------------------------------------------------------------
extern "C" void kernel_launch(void* const* d_in, const int* in_sizes, int n_in,
                              void* d_out, int out_size, void* d_ws, size_t ws_size,
                              hipStream_t stream)
{
  (void)in_sizes; (void)n_in; (void)out_size; (void)ws_size;

  const float* x_in  = (const float*)d_in[0];
  const float* aln_g = (const float*)d_in[1];
  const float* aln_b = (const float*)d_in[2];
  const float* qkv_w = (const float*)d_in[3];
  const float* out_w = (const float*)d_in[4];
  const float* fln_g = (const float*)d_in[5];
  const float* fln_b = (const float*)d_in[6];
  const float* ff_w1 = (const float*)d_in[7];
  const float* ff_b1 = (const float*)d_in[8];
  const float* ff_w2 = (const float*)d_in[9];
  const float* ff_b2 = (const float*)d_in[10];
  const float* gln_g = (const float*)d_in[11];
  const float* gln_b = (const float*)d_in[12];

  char* ws = (char*)d_ws;
  size_t off = 0;
  float*    xbuf  = (float*)(ws + off);    off += (size_t)NTOK * DIMD * 4;    // 16 MB
  ushort_t* xn    = (ushort_t*)(ws + off); off += (size_t)NTOK * DIMD * 2;    //  8 MB
  ushort_t* Qb    = (ushort_t*)(ws + off); off += (size_t)NTOK * INNERD * 2;  //  4 MB
  ushort_t* Kb    = (ushort_t*)(ws + off); off += (size_t)NTOK * INNERD * 2;
  ushort_t* Vb    = (ushort_t*)(ws + off); off += (size_t)NTOK * INNERD * 2;
  ushort_t* attno = (ushort_t*)(ws + off); off += (size_t)NTOK * INNERD * 2;
  float*    qkvf  = (float*)(ws + off);    // aliases hbuf region (disjoint lifetime)
  ushort_t* hbuf  = (ushort_t*)(ws + off); off += (size_t)NTOK * FFD * 2;     // 32 MB
  ushort_t* wbuf  = (ushort_t*)(ws + off); off += (size_t)DIMD * FFD * 2;     //  8 MB
  // total ~84 MB

  (void)hipMemcpyAsync(xbuf, x_in, (size_t)NTOK * DIMD * 4,
                       hipMemcpyDeviceToDevice, stream);

  const dim3 blk(256);
  for (int i = 0; i < 6; ++i) {
    // --- attention block ---
    ln_kernel<<<NTOK, blk, 0, stream>>>(xbuf, aln_g + i * DIMD, aln_b + i * DIMD,
                                        xn, nullptr, 1);
    cvt_kernel<<<(DIMD * 1536 / 4 + 255) / 256, blk, 0, stream>>>(
        qkv_w + (size_t)i * DIMD * 1536, wbuf, DIMD * 1536 / 4);
    gemm_bf16_kernel<<<dim3(1536 / BN, NTOK / BM), blk, 0, stream>>>(
        xn, wbuf, nullptr, nullptr, qkvf, nullptr, NTOK, 1536, DIMD, 0);
    repack_qkv_kernel<<<NTOK * 1536 / 256, blk, 0, stream>>>(qkvf, Qb, Kb, Vb);
    attn_kernel<<<dim3(SEQ / 128, 16), blk, 0, stream>>>(Qb, Kb, Vb, attno);
    cvt_kernel<<<(INNERD * DIMD / 4 + 255) / 256, blk, 0, stream>>>(
        out_w + (size_t)i * INNERD * DIMD, wbuf, INNERD * DIMD / 4);
    gemm_bf16_kernel<<<dim3(DIMD / BN, NTOK / BM), blk, 0, stream>>>(
        attno, wbuf, nullptr, xbuf, xbuf, nullptr, NTOK, DIMD, INNERD, 4);

    // --- FFN block ---
    ln_kernel<<<NTOK, blk, 0, stream>>>(xbuf, fln_g + i * DIMD, fln_b + i * DIMD,
                                        xn, nullptr, 1);
    cvt_kernel<<<(DIMD * FFD / 4 + 255) / 256, blk, 0, stream>>>(
        ff_w1 + (size_t)i * DIMD * FFD, wbuf, DIMD * FFD / 4);
    gemm_bf16_kernel<<<dim3(FFD / BN, NTOK / BM), blk, 0, stream>>>(
        xn, wbuf, ff_b1 + (size_t)i * FFD, nullptr, nullptr, hbuf,
        NTOK, FFD, DIMD, 1 | 2 | 8);
    cvt_kernel<<<(FFD * DIMD / 4 + 255) / 256, blk, 0, stream>>>(
        ff_w2 + (size_t)i * FFD * DIMD, wbuf, FFD * DIMD / 4);
    gemm_bf16_kernel<<<dim3(DIMD / BN, NTOK / BM), blk, 0, stream>>>(
        hbuf, wbuf, ff_b2 + (size_t)i * DIMD, xbuf, xbuf, nullptr,
        NTOK, DIMD, FFD, 1 | 4);
  }

  ln_kernel<<<NTOK, blk, 0, stream>>>(xbuf, gln_g, gln_b, nullptr, (float*)d_out, 0);
}
